// AdaptiveGraphLayer_57088705299181
// MI455X (gfx1250) — matrix operand order repaired and checked by reading, not compiled, not run on hardware
//
#include <hip/hip_runtime.h>
#include <hip/hip_bf16.h>

// ---------------------------------------------------------------------------
// AdaptiveGraphLayer for MI455X (gfx1250, wave32, WMMA).
// B=8, T=64, N=256, D=128, EMB=128.
// GEMMs run on v_wmma_f32_16x16x32_f16 (f16 operands, f32 accumulate);
// softmax / gate / elementwise / LayerNorm stay in f32.
// All WMMA fragments are aligned b128 vector loads; the dominant adj@x GEMM
// is register-blocked 32x32 per wave (4 WMMAs per 4 fragment loads).
// ---------------------------------------------------------------------------

typedef __attribute__((ext_vector_type(16))) _Float16 v16h;
typedef __attribute__((ext_vector_type(8)))  _Float16 v8h;
typedef __attribute__((ext_vector_type(4)))  _Float16 v4h;
typedef __attribute__((ext_vector_type(8)))  float    v8f;

#define BB 8
#define TT 64
#define NN 256
#define DD 128

// ---- WMMA fragment loaders (contiguous k order, same for A and B) -----
// A 16x32 f16: lane L holds row m = L&15; element i: k = (L>>4)*16 + i.
// For a row-major source (element (m,k) at p[m*ld+k]) this is two contiguous
// 16-byte chunks per lane: [g*16 .. g*16+7] and [g*16+8 .. g*16+15].
__device__ inline v16h load_a_frag_rm(const _Float16* p, int lane, int ld) {
  int m = lane & 15, g = lane >> 4;
  const _Float16* base = p + (size_t)m * ld + g * 16;
  v8h lo = *(const v8h*)(base);
  v8h hi = *(const v8h*)(base + 8);
  v16h a;
#pragma unroll
  for (int i = 0; i < 8; ++i) { a[i] = lo[i]; a[i + 8] = hi[i]; }
  return a;
}

// B 32x16 f16: lane L holds column n = L&15; element i: k = (L>>4)*16 + i.
// For a K-contiguous source (element (k,n) at p[n*ld + k]) this is one aligned
// 32-byte chunk per lane.
__device__ inline v16h load_b_frag_kc(const _Float16* p, int lane, int ld) {
  int n = lane & 15, g = lane >> 4;
  return *(const v16h*)(p + (size_t)n * ld + g * 16);
}

__device__ inline v8f wmma_f16(v16h a, v16h b, v8f c) {
  // (neg_a, A, neg_b, B, c_mod, C, reuse_a, reuse_b)
  return __builtin_amdgcn_wmma_f32_16x16x32_f16(false, a, false, b,
                                                (short)0, c, false, false);
}

// ---------------------------------------------------------------------------
// Kernel 1: adjacency = softmax(mask(emb1 @ emb2^T)) -> f16 [N,N]
// ---------------------------------------------------------------------------
__global__ void adj_kernel(const float* __restrict__ emb1,
                           const float* __restrict__ emb2,
                           _Float16* __restrict__ adj_h) {
  __shared__ float row[DD];
  __shared__ float red[NN];
  int m = blockIdx.x;
  int n = threadIdx.x;
  if (n < DD) row[n] = emb1[m * DD + n];
  __syncthreads();
  float dot = 0.f;
#pragma unroll 8
  for (int d = 0; d < DD; ++d) dot += row[d] * emb2[n * DD + d];
  float masked = (dot > 0.01f) ? dot : -1e9f;

  red[n] = masked;
  __syncthreads();
  for (int s = 128; s > 0; s >>= 1) {
    if (n < s) red[n] = fmaxf(red[n], red[n + s]);
    __syncthreads();
  }
  float mx = red[0];
  __syncthreads();
  float e = __expf(masked - mx);
  red[n] = e;
  __syncthreads();
  for (int s = 128; s > 0; s >>= 1) {
    if (n < s) red[n] += red[n + s];
    __syncthreads();
  }
  float sum = red[0];
  adj_h[m * NN + n] = (_Float16)(e / sum);
}

// ---------------------------------------------------------------------------
// Kernel 2: gate[b,n] = sigmoid(Wa2 @ relu(Wa1 @ mean_t(x[b,:,n,:]) + ba1) + ba2)
// ---------------------------------------------------------------------------
__global__ void gate_kernel(const float* __restrict__ x,
                            const float* __restrict__ Wa1,
                            const float* __restrict__ ba1,
                            const float* __restrict__ Wa2,
                            const float* __restrict__ ba2,
                            float* __restrict__ gate) {
  __shared__ float ctx[DD];
  __shared__ float h[DD / 4];
  int bn = blockIdx.x;
  int b = bn >> 8, n = bn & 255;
  int d = threadIdx.x;
  const float* xp = x + ((size_t)b * TT * NN + n) * DD + d;
  float s = 0.f;
  for (int t = 0; t < TT; ++t) s += xp[(size_t)t * NN * DD];
  ctx[d] = s * (1.f / (float)TT);
  __syncthreads();
  if (d < DD / 4) {
    float acc = ba1[d];
#pragma unroll 8
    for (int j = 0; j < DD; ++j) acc += ctx[j] * Wa1[d * DD + j];
    h[d] = fmaxf(acc, 0.f);
  }
  __syncthreads();
  if (d == 0) {
    float acc = ba2[0];
#pragma unroll
    for (int j = 0; j < DD / 4; ++j) acc += h[j] * Wa2[j];
    gate[bn] = 1.f / (1.f + __expf(-acc));
  }
}

// ---------------------------------------------------------------------------
// Kernel 3: x[b,t][n][d] (f32) -> xT[b,t][d][n] (f16), 32x32 LDS tile transpose.
// grid = (32 tiles, B*T), block = 256.
// ---------------------------------------------------------------------------
__global__ void xT_kernel(const float* __restrict__ x,
                          _Float16* __restrict__ xT) {
  __shared__ _Float16 tile[32][36];       // padded to dodge bank conflicts
  int t  = blockIdx.x;                    // 8 n-tiles x 4 d-tiles
  int bt = blockIdx.y;
  int nt = t >> 2, dt = t & 3;
  int tid = threadIdx.x;

  int nl  = tid >> 3;                     // 0..31 row within n-tile
  int dgi = (tid & 7) * 4;                // 0..28 col group within d-tile
  const float* src =
      x + ((size_t)bt * NN + nt * 32 + nl) * DD + dt * 32 + dgi;
  float4 v = *(const float4*)src;
  tile[nl][dgi + 0] = (_Float16)v.x;
  tile[nl][dgi + 1] = (_Float16)v.y;
  tile[nl][dgi + 2] = (_Float16)v.z;
  tile[nl][dgi + 3] = (_Float16)v.w;
  __syncthreads();

  int dl  = tid >> 3;                     // 0..31 row within d-tile (output)
  int ngi = (tid & 7) * 4;                // 0..28 col group within n-tile
  v4h o;
  o[0] = tile[ngi + 0][dl];
  o[1] = tile[ngi + 1][dl];
  o[2] = tile[ngi + 2][dl];
  o[3] = tile[ngi + 3][dl];
  _Float16* dst = xT + (size_t)bt * DD * NN +
                  (size_t)(dt * 32 + dl) * NN + nt * 32 + ngi;
  *(v4h*)dst = o;
}

// ---------------------------------------------------------------------------
// Kernel 4: weights f32 -> f16 (layout unchanged, already K-contiguous).
// ---------------------------------------------------------------------------
__global__ void wcvt_kernel(const float* __restrict__ Wa,
                            const float* __restrict__ Wm,
                            const float* __restrict__ Wo,
                            _Float16* __restrict__ Wah,
                            _Float16* __restrict__ Wmh,
                            _Float16* __restrict__ Woh) {
  int i = blockIdx.x * blockDim.x + threadIdx.x;
  if (i < DD * DD) { Wah[i] = (_Float16)Wa[i]; Wmh[i] = (_Float16)Wm[i]; }
  if (i < DD * 2 * DD) Woh[i] = (_Float16)Wo[i];
}

// ---------------------------------------------------------------------------
// Kernel 5: aggr[b,t] = diag(gate[b,:]) * (adj @ x[b,t])  -> f16 [N,D] row-major
// One wave per 32x32 macro-tile: 2 A-frags x 2 B-frags -> 4 WMMAs per k-step
// (1.0 fragment loads per WMMA). K = 256 (8 k-steps, 32 WMMAs).
// grid = (32 macro-tiles, B*T), block = 32.
// ---------------------------------------------------------------------------
__global__ void aggr_kernel(const _Float16* __restrict__ adj_h,
                            const _Float16* __restrict__ xT,
                            const float* __restrict__ gate,
                            _Float16* __restrict__ aggr_h) {
  int tile = blockIdx.x;            // 8 m-pairs x 4 n-pairs
  int bt   = blockIdx.y;
  int mp = tile >> 2, np = tile & 3;
  int b  = bt >> 6;
  int lane = threadIdx.x;           // blockDim.x == 32, EXEC all ones

  const _Float16* adj0 = adj_h + (size_t)(mp * 32) * NN;        // adj[m,k]
  const _Float16* adj1 = adj0 + (size_t)16 * NN;
  const _Float16* xp0  = xT + (size_t)bt * DD * NN +
                         (size_t)(np * 32) * NN;                // xT[d=n][k]
  const _Float16* xp1  = xp0 + (size_t)16 * NN;

  v8f c00 = {}, c01 = {}, c10 = {}, c11 = {};
#pragma unroll
  for (int ks = 0; ks < NN; ks += 32) {
    v16h a0 = load_a_frag_rm(adj0 + ks, lane, NN);
    v16h a1 = load_a_frag_rm(adj1 + ks, lane, NN);
    v16h b0 = load_b_frag_kc(xp0 + ks, lane, NN);
    v16h b1 = load_b_frag_kc(xp1 + ks, lane, NN);
    c00 = wmma_f16(a0, b0, c00);
    c01 = wmma_f16(a0, b1, c01);
    c10 = wmma_f16(a1, b0, c10);
    c11 = wmma_f16(a1, b1, c11);
  }

  int nn = lane & 15, g = lane >> 4;
  _Float16* outp = aggr_h + (size_t)bt * NN * DD;
#pragma unroll
  for (int r = 0; r < 8; ++r) {
    int row0 = mp * 32 + r + 8 * g;       // D layout: lanes 16-31 carry M+8
    int row1 = row0 + 16;
    float g0 = gate[b * NN + row0];
    float g1 = gate[b * NN + row1];
    int col0 = np * 32 + nn;
    int col1 = col0 + 16;
    outp[(size_t)row0 * DD + col0] = (_Float16)(c00[r] * g0);
    outp[(size_t)row0 * DD + col1] = (_Float16)(c01[r] * g0);
    outp[(size_t)row1 * DD + col0] = (_Float16)(c10[r] * g1);
    outp[(size_t)row1 * DD + col1] = (_Float16)(c11[r] * g1);
  }
}

// ---------------------------------------------------------------------------
// Kernel 6 (fused): per 16-row block of one (b,t):
//   S = aggr @ W_add^T + b_add                (f16, LDS)
//   R = (aggr @ W_mul^T + b_mul) * x          (f16, LDS)
//   out = [S|R] @ W_out^T + b_out             (K=256, A from LDS)
//   y = x + out ; LayerNorm(y) * gamma + beta (f32 -> HBM)
// grid = (N/16, B*T), block = 256 (8 waves; wave w owns column tile w).
// ---------------------------------------------------------------------------
__global__ void out_kernel(const _Float16* __restrict__ aggr_h,
                           const float* __restrict__ x,
                           const _Float16* __restrict__ WaddH,
                           const float* __restrict__ b_add,
                           const _Float16* __restrict__ WmulH,
                           const float* __restrict__ b_mul,
                           const _Float16* __restrict__ WoutH,
                           const float* __restrict__ b_out,
                           const float* __restrict__ gamma,
                           const float* __restrict__ beta,
                           float* __restrict__ out) {
  __shared__ __align__(32) _Float16 Sh[16][DD];
  __shared__ __align__(32) _Float16 Rh[16][DD];
  __shared__ float Y[16][DD];
  __shared__ float ps1[16][16], ps2[16][16];
  __shared__ float mean_s[16], inv_s[16];

  int mb = blockIdx.x;              // 16-row block within N
  int bt = blockIdx.y;
  int tid = threadIdx.x;
  int w = tid >> 5;                 // wave id = output column tile
  int lane = tid & 31;

  const _Float16* ag = aggr_h + (size_t)bt * NN * DD + (size_t)mb * 16 * DD;
  const float* xrow  = x + ((size_t)bt * NN + mb * 16) * DD;

  // ---- stage 1: branch projections (K = 128), A reused for both B's ----
  v8f cP = {}, cQ = {};
#pragma unroll
  for (int ks = 0; ks < DD; ks += 32) {
    v16h a  = load_a_frag_rm(ag + ks, lane, DD);                // aggr[m,k]
    v16h bP = load_b_frag_kc(WaddH + (size_t)w * 16 * DD + ks, lane, DD);
    v16h bQ = load_b_frag_kc(WmulH + (size_t)w * 16 * DD + ks, lane, DD);
    cP = wmma_f16(a, bP, cP);
    cQ = wmma_f16(a, bQ, cQ);
  }
  int nn = lane & 15, g = lane >> 4;
  int col = w * 16 + nn;
#pragma unroll
  for (int r = 0; r < 8; ++r) {
    int row = r + 8 * g;
    float p = cP[r] + b_add[col];
    float q = (cQ[r] + b_mul[col]) * xrow[(size_t)row * DD + col];
    Sh[row][col] = (_Float16)p;
    Rh[row][col] = (_Float16)q;
  }
  __syncthreads();

  // ---- stage 2: out = [S|R] @ W_out^T (K = 256, A from LDS) ----
  v8f c = {};
#pragma unroll
  for (int ks = 0; ks < 2 * DD; ks += 32) {
    const _Float16* src = (ks < DD) ? &Sh[0][0] : &Rh[0][0];
    int kk = (ks < DD) ? ks : ks - DD;
    v16h a  = load_a_frag_rm(src + kk, lane, DD);
    v16h bW = load_b_frag_kc(WoutH + (size_t)w * 16 * (2 * DD) + ks, lane, 2 * DD);
    c = wmma_f16(a, bW, c);
  }
#pragma unroll
  for (int r = 0; r < 8; ++r) {
    int row = r + 8 * g;
    Y[row][col] = c[r] + b_out[col] + xrow[(size_t)row * DD + col];
  }
  __syncthreads();

  // ---- stage 3: LayerNorm over D per row ----
  int rrow = tid >> 4, l16 = tid & 15;
  float s1 = 0.f, s2 = 0.f;
#pragma unroll
  for (int cc = l16; cc < DD; cc += 16) {
    float v = Y[rrow][cc];
    s1 += v; s2 += v * v;
  }
  ps1[rrow][l16] = s1;
  ps2[rrow][l16] = s2;
  __syncthreads();
  if (l16 == 0) {
    float a1 = 0.f, a2 = 0.f;
#pragma unroll
    for (int j = 0; j < 16; ++j) { a1 += ps1[rrow][j]; a2 += ps2[rrow][j]; }
    float mu  = a1 * (1.f / (float)DD);
    float var = a2 * (1.f / (float)DD) - mu * mu;
    mean_s[rrow] = mu;
    inv_s[rrow]  = rsqrtf(var + 1e-5f);
  }
  __syncthreads();
  float* op = out + ((size_t)bt * NN + mb * 16) * DD;
#pragma unroll
  for (int cc = l16; cc < DD; cc += 16) {
    float v = (Y[rrow][cc] - mean_s[rrow]) * inv_s[rrow];
    op[(size_t)rrow * DD + cc] = gamma[cc] * v + beta[cc];
  }
}

// ---------------------------------------------------------------------------
// Launcher
// ---------------------------------------------------------------------------
extern "C" void kernel_launch(void* const* d_in, const int* in_sizes, int n_in,
                              void* d_out, int out_size, void* d_ws, size_t ws_size,
                              hipStream_t stream) {
  const float* x     = (const float*)d_in[0];
  const float* emb1  = (const float*)d_in[1];
  const float* emb2  = (const float*)d_in[2];
  const float* W_add = (const float*)d_in[3];
  const float* b_add = (const float*)d_in[4];
  const float* W_mul = (const float*)d_in[5];
  const float* b_mul = (const float*)d_in[6];
  const float* Wa1   = (const float*)d_in[7];
  const float* ba1   = (const float*)d_in[8];
  const float* Wa2   = (const float*)d_in[9];
  const float* ba2   = (const float*)d_in[10];
  const float* W_out = (const float*)d_in[11];
  const float* b_out = (const float*)d_in[12];
  const float* gamma = (const float*)d_in[13];
  const float* beta  = (const float*)d_in[14];
  float* out = (float*)d_out;

  // Workspace layout (bytes, all 32B-aligned):
  //   adj_h  @ 0        : N*N*2        = 131072
  //   gate   @ 131072   : B*N*4        =   8192
  //   WaddH  @ 139264   : D*D*2        =  32768
  //   WmulH  @ 172032   : D*D*2        =  32768
  //   WoutH  @ 204800   : D*2D*2       =  65536
  //   xT     @ 270336   : B*T*D*N*2    = 33554432
  //   aggr_h @ 33824768 : B*T*N*D*2    = 33554432   (end 67379200)
  char* ws = (char*)d_ws;
  _Float16* adj_h  = (_Float16*)(ws);
  float*    gate   = (float*)   (ws + 131072);
  _Float16* WaddH  = (_Float16*)(ws + 139264);
  _Float16* WmulH  = (_Float16*)(ws + 172032);
  _Float16* WoutH  = (_Float16*)(ws + 204800);
  _Float16* xT     = (_Float16*)(ws + 270336);
  _Float16* aggr_h = (_Float16*)(ws + 33824768);

  adj_kernel <<<NN, NN, 0, stream>>>(emb1, emb2, adj_h);
  gate_kernel<<<BB * NN, DD, 0, stream>>>(x, Wa1, ba1, Wa2, ba2, gate);
  xT_kernel  <<<dim3(32, BB * TT), 256, 0, stream>>>(x, xT);
  wcvt_kernel<<<(DD * 2 * DD + 255) / 256, 256, 0, stream>>>(
      W_add, W_mul, W_out, WaddH, WmulH, WoutH);
  aggr_kernel<<<dim3(32, BB * TT), 32, 0, stream>>>(adj_h, xT, gate, aggr_h);
  out_kernel <<<dim3(NN / 16, BB * TT), 256, 0, stream>>>(
      aggr_h, x, WaddH, b_add, WmulH, b_mul, WoutH, b_out, gamma, beta, out);
}
